// TorchLSTM_88716844466425
// MI455X (gfx1250) — matrix operand order repaired and checked by reading, not compiled, not run on hardware
//
#include <hip/hip_runtime.h>
#include <hip/hip_bf16.h>
#include <cstdint>

// Problem dims (from reference)
#define T_DIM   64
#define SEQ_DIM 128          // batch for the scan
#define IN_DIM  256
#define H_DIM   512
#define NC_DIM  10
#define G4H     (4 * H_DIM)  // 2048
#define TB      (T_DIM * SEQ_DIM) // 8192
#define LDS_STRIDE (H_DIM + 8)   // +8 bf16 pad -> +4 banks/row, breaks conflicts

typedef __bf16 bf16_t;
typedef bf16_t v16bf __attribute__((ext_vector_type(16)));
typedef float  v8f   __attribute__((ext_vector_type(8)));

union FragU { v16bf v; uint32_t u[8]; };

// ---- WMMA fragment loaders (contiguous K order, A and B match), pointer-hoisted --------
// prow for A = base + row*stride + (lane<16 ? 0 : 16)  (row = lane&15)
// lanes 0-15: V0..3 K=k0+0..7,   V4..7 K=k0+8..15
// lanes16-31: V0..3 K=k0+16..23, V4..7 K=k0+24..31
__device__ __forceinline__ v16bf load_a_frag_p(const bf16_t* prow, int k0) {
    const bf16_t* p = prow + k0;
    FragU f;
#pragma unroll
    for (int v = 0; v < 4; ++v) {
        f.u[v]     = *(const uint32_t*)(p + 2 * v);
        f.u[v + 4] = *(const uint32_t*)(p + 8 + 2 * v);
    }
    return f.v;
}

// prow for B = W + n*ldw + (lane<16 ? 0 : 16)    (n = nbase + (lane&15))
// lanes 0-15: Vv K=k0+2v,2v+1 ; lanes16-31: Vv K=k0+16+2v,+1
__device__ __forceinline__ v16bf load_b_frag_p(const bf16_t* prow, int k0) {
    const bf16_t* p = prow + k0;
    FragU f;
#pragma unroll
    for (int v = 0; v < 8; ++v) f.u[v] = *(const uint32_t*)(p + 2 * v);
    return f.v;
}

__device__ __forceinline__ v8f wmma_bf16(v16bf a, v16bf b, v8f c) {
    return __builtin_amdgcn_wmma_f32_16x16x32_bf16(
        /*neg_a=*/false, a, /*neg_b=*/false, b,
        /*c_mod=*/(short)0, c, /*reuse_a=*/false, /*reuse_b=*/false);
}

// ---- Utility kernels ----------------------------------------------------
__global__ void cvt_f32_bf16(const float* __restrict__ src,
                             bf16_t* __restrict__ dst, int n) {
    int i = blockIdx.x * blockDim.x + threadIdx.x;
    int stride = gridDim.x * blockDim.x;
    for (; i < n; i += stride) dst[i] = (bf16_t)src[i];
}

// ---- Generic bf16 WMMA GEMM: C = A[M,K] * W[N,K]^T (+bias, relu) --------
// One wave computes a 16(M) x 64(N) tile. kchunks>1: split-K, each
// (kc,mt,ng) wave stores its partial tile to a unique slot (deterministic).
__global__ __launch_bounds__(256) void gemm_bf16_nt(
    const bf16_t* __restrict__ A, const bf16_t* __restrict__ W,
    const float* __restrict__ bias, float* __restrict__ C,
    int M, int N, int K, int relu, int kchunks)
{
    const int lane = threadIdx.x & 31;
    const int wid  = (int)((blockIdx.x * blockDim.x + threadIdx.x) >> 5);
    const int Mt = M >> 4;
    const int Ng = N >> 6;
    const int tilesPer = Mt * Ng;
    if (wid >= tilesPer * kchunks) return;   // wave-uniform exit
    const int kc  = wid / tilesPer;
    const int rem = wid - kc * tilesPer;
    const int mt  = rem / Ng;
    const int ng  = rem - mt * Ng;
    const int mbase = mt << 4;
    const int nbase = ng << 6;
    const int col  = lane & 15;
    const int kper = K / kchunks;
    const int kbeg = kc * kper;
    const int kend = kbeg + kper;

    const bf16_t* pa = A + (size_t)(mbase + col) * (size_t)K + ((lane < 16) ? 0 : 16);
    const int kbB = (lane < 16) ? 0 : 16;
    const bf16_t* pb0 = W + (size_t)(nbase +  0 + col) * (size_t)K + kbB;
    const bf16_t* pb1 = W + (size_t)(nbase + 16 + col) * (size_t)K + kbB;
    const bf16_t* pb2 = W + (size_t)(nbase + 32 + col) * (size_t)K + kbB;
    const bf16_t* pb3 = W + (size_t)(nbase + 48 + col) * (size_t)K + kbB;

    v8f a0 = {}, a1 = {}, a2 = {}, a3 = {};
    for (int k0 = kbeg; k0 < kend; k0 += 32) {
        v16bf a = load_a_frag_p(pa, k0);
        a0 = wmma_bf16(a, load_b_frag_p(pb0, k0), a0);
        a1 = wmma_bf16(a, load_b_frag_p(pb1, k0), a1);
        a2 = wmma_bf16(a, load_b_frag_p(pb2, k0), a2);
        a3 = wmma_bf16(a, load_b_frag_p(pb3, k0), a3);
    }

    float* Cout = (kchunks == 1) ? C : (C + (size_t)kc * (size_t)M * (size_t)N);
    const int mrow = mbase + ((lane >> 4) << 3);   // C/D layout: M = r + 8*laneHi
    v8f accs[4] = {a0, a1, a2, a3};
#pragma unroll
    for (int s = 0; s < 4; ++s) {
        const int n = nbase + 16 * s + col;
        const float bv = (kchunks == 1 && bias) ? bias[n] : 0.0f;
#pragma unroll
        for (int r = 0; r < 8; ++r) {
            float v = accs[s][r] + bv;
            if (relu && kchunks == 1) v = fmaxf(v, 0.0f);
            Cout[(size_t)(mrow + r) * (size_t)N + (size_t)n] = v;
        }
    }
}

// Deterministic split-K reduction + bias + ReLU
__global__ void splitk_reduce_bias_relu(const float* __restrict__ part,
                                        const float* __restrict__ bias,
                                        float* __restrict__ C,
                                        int MN, int N, int kchunks) {
    int i = blockIdx.x * blockDim.x + threadIdx.x;
    if (i >= MN) return;
    float s = bias ? bias[i % N] : 0.0f;
    for (int kc = 0; kc < kchunks; ++kc) s += part[(size_t)kc * MN + i];
    C[i] = fmaxf(s, 0.0f);
}

// ---- Persistent LSTM layer, one workgroup per 16-batch-row stripe -------
// Block = 1024 threads = 32 waves (one per 16-col N-tile of H). Each block
// owns batch rows [16*mt, 16*mt+16) for the ENTIRE 64-step scan: the
// recurrent h tile (16 x 512 bf16) ping-pongs through LDS (double
// buffered), so inter-step sync is a plain __syncthreads() -- no global
// barrier, no fences. Cell state c lives in 8 registers per lane. Each
// wave computes all FOUR gate tiles (n, H+n, 2H+n, 3H+n) so the
// nonlinearities fuse in-register.
__global__ __launch_bounds__(1024) void lstm_layer_persistent(
    const float*  __restrict__ xg,     // [T, B, 4H]  (x @ w_ih^T + b_ih)
    const bf16_t* __restrict__ whh,    // [4H, H] bf16
    const float*  __restrict__ bhh,    // [4H]
    bf16_t*       __restrict__ hseq)   // [T, B, H] out (bf16 for next GEMM)
{
    __shared__ bf16_t hbuf[2][16 * LDS_STRIDE];   // 2 x 16.25KB ping-pong

    const int lane = threadIdx.x & 31;
    const int nt   = (int)(threadIdx.x >> 5);     // 0..31 N-tile
    const int mt   = (int)blockIdx.x;             // 0..7  batch stripe
    const int mbase = mt << 4;
    const int nbase = nt << 4;
    const int col  = lane & 15;
    const int j    = nbase + col;                 // hidden column
    const int mrowL = (lane >> 4) << 3;           // 0 or 8 (row in tile)

    // hoisted B-row base pointers (4 gates), K-offset folded per lane-half
    const int kbB = (lane < 16) ? 0 : 16;
    const bf16_t* pbi = whh + (size_t)(0 * H_DIM + j) * H_DIM + kbB;
    const bf16_t* pbf = whh + (size_t)(1 * H_DIM + j) * H_DIM + kbB;
    const bf16_t* pbg = whh + (size_t)(2 * H_DIM + j) * H_DIM + kbB;
    const bf16_t* pbo = whh + (size_t)(3 * H_DIM + j) * H_DIM + kbB;
    // LDS A row pointer: row-in-tile = lane&15, K-offset per lane-half
    const int kbA = (lane < 16) ? 0 : 16;
    const int aoffL = col * LDS_STRIDE + kbA;

    // biases are loop-invariant: load once
    const float bi = bhh[j];
    const float bf = bhh[H_DIM + j];
    const float bg = bhh[2 * H_DIM + j];
    const float bo = bhh[3 * H_DIM + j];

    float creg[8];
#pragma unroll
    for (int r = 0; r < 8; ++r) creg[r] = 0.0f;

    for (int t = 0; t < T_DIM; ++t) {
        const int pb = t & 1;                     // read buf, write 1-pb
        v8f ai = {}, af = {}, ag = {}, ao = {};
        if (t > 0) {   // uniform branch: EXEC stays all-ones for WMMA
            const bf16_t* pa = &hbuf[pb][aoffL];  // ds_load A-fragments
            for (int k0 = 0; k0 < H_DIM; k0 += 32) {
                v16bf a = load_a_frag_p(pa, k0);
                ai = wmma_bf16(a, load_b_frag_p(pbi, k0), ai);
                af = wmma_bf16(a, load_b_frag_p(pbf, k0), af);
                ag = wmma_bf16(a, load_b_frag_p(pbg, k0), ag);
                ao = wmma_bf16(a, load_b_frag_p(pbo, k0), ao);
            }
        }

        const float* xgt = xg + (size_t)t * SEQ_DIM * G4H;
        bf16_t* hout = hseq + (size_t)t * SEQ_DIM * H_DIM;
        bf16_t* hnext = &hbuf[1 - pb][0];
#pragma unroll
        for (int r = 0; r < 8; ++r) {
            const int rowT = mrowL + r;           // 0..15 within stripe
            const int b = mbase + rowT;           // global batch row
            const float* xr = xgt + (size_t)b * G4H;
            float gi = ai[r] + xr[j]             + bi;
            float gf = af[r] + xr[H_DIM + j]     + bf;
            float gg = ag[r] + xr[2 * H_DIM + j] + bg;
            float go = ao[r] + xr[3 * H_DIM + j] + bo;
            gi = 1.0f / (1.0f + __expf(-gi));
            gf = 1.0f / (1.0f + __expf(-gf));
            gg = tanhf(gg);
            go = 1.0f / (1.0f + __expf(-go));
            float cn = gf * creg[r] + gi * gg;
            creg[r] = cn;
            const bf16_t hv = (bf16_t)(go * tanhf(cn));
            hnext[rowT * LDS_STRIDE + j] = hv;            // for the recurrence
            hout[(size_t)b * H_DIM + (size_t)j] = hv;     // for later GEMMs
        }

        __syncthreads();   // h(t) LDS writes visible before step t+1 reads
    }
}

// ---- Tiny final Linear(H -> NC), fp32 -----------------------------------
__global__ void linear_out(const float* __restrict__ hmid,
                           const float* __restrict__ w2,
                           const float* __restrict__ b2,
                           float* __restrict__ out) {
    int idx = blockIdx.x * blockDim.x + threadIdx.x;
    if (idx >= T_DIM * NC_DIM) return;
    int t = idx / NC_DIM, n = idx % NC_DIM;
    float s = b2[n];
    for (int k = 0; k < H_DIM; ++k)
        s += hmid[(size_t)t * H_DIM + k] * w2[(size_t)n * H_DIM + k];
    out[idx] = s;
}

// ---- Host orchestration -------------------------------------------------
extern "C" void kernel_launch(void* const* d_in, const int* in_sizes, int n_in,
                              void* d_out, int out_size, void* d_ws, size_t ws_size,
                              hipStream_t stream) {
    (void)in_sizes; (void)n_in; (void)out_size; (void)ws_size;

    const float* x     = (const float*)d_in[0];
    const float* w_ih0 = (const float*)d_in[1];
    const float* w_hh0 = (const float*)d_in[2];
    const float* b_ih0 = (const float*)d_in[3];
    const float* b_hh0 = (const float*)d_in[4];
    const float* w_ih1 = (const float*)d_in[5];
    const float* w_hh1 = (const float*)d_in[6];
    const float* b_ih1 = (const float*)d_in[7];
    const float* b_hh1 = (const float*)d_in[8];
    const float* w1    = (const float*)d_in[9];
    const float* b1    = (const float*)d_in[10];
    const float* w2    = (const float*)d_in[11];
    const float* b2    = (const float*)d_in[12];

    char* ws = (char*)d_ws;
    size_t off = 0;
    auto alloc = [&](size_t bytes) -> void* {
        void* p = ws + off;
        off = (off + bytes + 255) & ~(size_t)255;
        return p;
    };
    float*  xg    = (float*) alloc((size_t)TB * G4H * 4);            // 64 MB (reused L0/L1)
    bf16_t* hseq0 = (bf16_t*)alloc((size_t)TB * H_DIM * 2);          // 8 MB
    bf16_t* hseq1 = (bf16_t*)alloc((size_t)TB * H_DIM * 2);          // 8 MB
    bf16_t* xbf   = (bf16_t*)alloc((size_t)TB * IN_DIM * 2);         // 4 MB
    bf16_t* wih0b = (bf16_t*)alloc((size_t)G4H * IN_DIM * 2);
    bf16_t* whh0b = (bf16_t*)alloc((size_t)G4H * H_DIM * 2);
    bf16_t* wih1b = (bf16_t*)alloc((size_t)G4H * H_DIM * 2);
    bf16_t* whh1b = (bf16_t*)alloc((size_t)G4H * H_DIM * 2);
    bf16_t* w1b   = (bf16_t*)alloc((size_t)H_DIM * SEQ_DIM * H_DIM * 2); // 64 MB, fits L2
    float*  hmid  = (float*) alloc((size_t)T_DIM * H_DIM * 4);
    float*  part  = (float*) alloc((size_t)64 * T_DIM * H_DIM * 4);  // split-K partials

    auto cvt = [&](const float* s, bf16_t* d, int n) {
        int blocks = (n + 255) / 256; if (blocks > 8192) blocks = 8192;
        cvt_f32_bf16<<<blocks, 256, 0, stream>>>(s, d, n);
    };
    cvt(x,     xbf,   TB * IN_DIM);
    cvt(w_ih0, wih0b, G4H * IN_DIM);
    cvt(w_hh0, whh0b, G4H * H_DIM);
    cvt(w_ih1, wih1b, G4H * H_DIM);
    cvt(w_hh1, whh1b, G4H * H_DIM);
    cvt(w1,    w1b,   H_DIM * SEQ_DIM * H_DIM);

    auto gemm_grid = [](int M, int N, int kchunks) {
        return ((M / 16) * (N / 64) * kchunks * 32 + 255) / 256;
    };

    // ---- Layer 0: xg = x @ w_ih0^T + b_ih0 (all timesteps at once) ----
    gemm_bf16_nt<<<gemm_grid(TB, G4H, 1), 256, 0, stream>>>(
        xbf, wih0b, b_ih0, xg, TB, G4H, IN_DIM, /*relu=*/0, /*kchunks=*/1);
    lstm_layer_persistent<<<8, 1024, 0, stream>>>(xg, whh0b, b_hh0, hseq0);

    // ---- Layer 1: xg = hseq0 @ w_ih1^T + b_ih1 ----
    gemm_bf16_nt<<<gemm_grid(TB, G4H, 1), 256, 0, stream>>>(
        hseq0, wih1b, b_ih1, xg, TB, G4H, H_DIM, 0, 1);
    lstm_layer_persistent<<<8, 1024, 0, stream>>>(xg, whh1b, b_hh1, hseq1);

    // ---- Head: hmid = relu([T, SEQ*H] @ w1^T + b1), split-K=64 ----
    const int KBIG = SEQ_DIM * H_DIM;      // 65536
    gemm_bf16_nt<<<gemm_grid(T_DIM, H_DIM, 64), 256, 0, stream>>>(
        hseq1, w1b, nullptr, part, T_DIM, H_DIM, KBIG, 0, /*kchunks=*/64);
    splitk_reduce_bias_relu<<<(T_DIM * H_DIM + 255) / 256, 256, 0, stream>>>(
        part, b1, hmid, T_DIM * H_DIM, H_DIM, 64);

    // ---- out = hmid @ w2^T + b2 ----
    linear_out<<<(T_DIM * NC_DIM + 255) / 256, 256, 0, stream>>>(
        hmid, w2, b2, (float*)d_out);
}
